// RNNModel_12979391168901
// MI455X (gfx1250) — matrix operand order repaired and checked by reading, not compiled, not run on hardware
//
#include <hip/hip_runtime.h>

typedef __attribute__((ext_vector_type(16))) __bf16 v16bf;
typedef __attribute__((ext_vector_type(8)))  __bf16 v8bf;
typedef __attribute__((ext_vector_type(8)))  float  v8f;

#define B_    256
#define T_    4096
#define DIN   64
#define DH    128
#define DOUT  32

#define XPITCH 72    // 64 + 8 halves (keeps 16B alignment, breaks bank conflicts)
#define HPITCH 136   // 128 + 8 halves

// fp32 -> bf16, round-to-nearest-even
__device__ __forceinline__ __bf16 f2bf(float f) {
    unsigned u = __float_as_uint(f);
    u += 0x7FFFu + ((u >> 16) & 1u);
    unsigned short h = (unsigned short)(u >> 16);
    __bf16 b;
    __builtin_memcpy(&b, &h, 2);
    return b;
}

__device__ __forceinline__ unsigned short f2bfu(float f) {
    unsigned u = __float_as_uint(f);
    u += 0x7FFFu + ((u >> 16) & 1u);
    return (unsigned short)(u >> 16);
}

// Build a 16-bit A/B fragment from two 16-byte LDS chunks (ds_load_b128 x2).
__device__ __forceinline__ v16bf frag_from_lds(const unsigned short* base, int off0) {
    v8bf lo = *(const v8bf*)(base + off0);
    v8bf hi = *(const v8bf*)(base + off0 + 16);
    v16bf r;
#pragma unroll
    for (int j = 0; j < 8; ++j) { r[j] = lo[j]; r[j + 8] = hi[j]; }
    return r;
}

// B-matrix fragment (K x 16 slice) built directly from global fp32 weights.
// Lane l supplies column N = row, K = k0 + (l>>4)*8 + (j&7) + (j>>3)*16  (two runs of 8 floats).
__device__ __forceinline__ v16bf bfrag_from_global(const float* W, int row, int rowlen,
                                                   int k0, int lane) {
    const float* p = W + (size_t)row * rowlen + k0 + ((lane >> 4) << 3);
    v16bf r;
#pragma unroll
    for (int j = 0; j < 16; ++j) r[j] = f2bf(p[(j & 7) + ((j >> 3) << 4)]);
    return r;
}

__global__ __launch_bounds__(256) void rnn_scan_wmma(
    const float* __restrict__ xs,   const float* __restrict__ W_ih,
    const float* __restrict__ b_ih, const float* __restrict__ W_hh,
    const float* __restrict__ b_hh, const float* __restrict__ W_out,
    const float* __restrict__ b_out, float* __restrict__ out)
{
    __shared__ __align__(32) unsigned short s_x[2][16 * XPITCH];  // staged xs tile (bf16)
    __shared__ __align__(32) unsigned short s_h[2][16 * HPITCH];  // double-buffered h (bf16)

    const int tid  = threadIdx.x;
    const int lane = tid & 31;
    const int wave = tid >> 5;                  // 0..7  -> h columns [wave*16, wave*16+16)
    const int b0   = blockIdx.x * 16;           // batch tile
    const int nrow = wave * 16 + (lane & 15);   // weight row this lane feeds into B frags
    const int mrow = lane & 15;                 // A-matrix M row this lane reads
    const int ksub = (lane >> 4) * 8;           // A-matrix K sub-offset per lane half

    // ---- per-wave constant weights as B-matrix fragments, kept in VGPRs ----
    v16bf Bih[2];
#pragma unroll
    for (int kk = 0; kk < 2; ++kk) Bih[kk] = bfrag_from_global(W_ih, nrow, DIN, kk * 32, lane);
    v16bf Bhh[4];
#pragma unroll
    for (int kk = 0; kk < 4; ++kk) Bhh[kk] = bfrag_from_global(W_hh, nrow, DH, kk * 32, lane);

    const float bias = b_ih[nrow] + b_hh[nrow];

    // ---- h_{-1} = 0 ; stage xs(t=0) ----
    for (int i = tid; i < 16 * HPITCH; i += 256) s_h[0][i] = 0;
    {
        const int m  = tid >> 4;
        const int i4 = (tid & 15) * 4;
        const float4 v = *(const float4*)(xs + ((size_t)(b0 + m) * T_) * DIN + i4);
        unsigned short* dst = &s_x[0][m * XPITCH + i4];
        dst[0] = f2bfu(v.x); dst[1] = f2bfu(v.y); dst[2] = f2bfu(v.z); dst[3] = f2bfu(v.w);
    }
    __syncthreads();

    for (int t = 0; t < T_; ++t) {
        const int pr = t & 1;
        const int pw = pr ^ 1;

        // Stage xs for t+1 into the other buffer while computing step t.
        if (t + 1 < T_) {
            const int m  = tid >> 4;
            const int i4 = (tid & 15) * 4;
            const float* src = xs + ((size_t)(b0 + m) * T_ + (t + 1)) * DIN + i4;
            __builtin_prefetch(src + DIN, 0, 0);  // pull xs(t+2) toward L2/L0
            const float4 v = *(const float4*)src;
            unsigned short* dst = &s_x[pw][m * XPITCH + i4];
            dst[0] = f2bfu(v.x); dst[1] = f2bfu(v.y); dst[2] = f2bfu(v.z); dst[3] = f2bfu(v.w);
        }

        // Two independent accumulator chains to shorten WMMA RAW chains.
        v8f accx = {};
#pragma unroll
        for (int kk = 0; kk < 2; ++kk) {
            v16bf a = frag_from_lds(s_x[pr], mrow * XPITCH + kk * 32 + ksub);
            accx = __builtin_amdgcn_wmma_f32_16x16x32_bf16(false, a, false, Bih[kk],
                                                           (short)0, accx, false, false);
        }
        v8f acch = {};
#pragma unroll
        for (int kk = 0; kk < 4; ++kk) {
            v16bf a = frag_from_lds(s_h[pr], mrow * HPITCH + kk * 32 + ksub);
            acch = __builtin_amdgcn_wmma_f32_16x16x32_bf16(false, a, false, Bhh[kk],
                                                           (short)0, acch, false, false);
        }

        // bias + hardtanh, publish h_t as bf16 A-matrix for next step
        unsigned short* hw = s_h[pw];
#pragma unroll
        for (int r = 0; r < 8; ++r) {
            float v = fminf(fmaxf(accx[r] + acch[r] + bias, -1.0f), 1.0f);
            hw[(r + ((lane >> 4) << 3)) * HPITCH + nrow] = f2bfu(v);
        }
        __syncthreads();
    }

    // ---- epilogue: out = h_last @ W_out^T + b_out ; h_last in s_h[T_&1] ----
    if (wave < 2) {
        v8f acc = {};
#pragma unroll
        for (int kk = 0; kk < 4; ++kk) {
            v16bf a = frag_from_lds(s_h[T_ & 1], mrow * HPITCH + kk * 32 + ksub);
            v16bf b = bfrag_from_global(W_out, nrow, DH, kk * 32, lane);
            acc = __builtin_amdgcn_wmma_f32_16x16x32_bf16(false, a, false, b,
                                                          (short)0, acc, false, false);
        }
        const float bo = b_out[nrow];
#pragma unroll
        for (int r = 0; r < 8; ++r) {
            const int m = r + ((lane >> 4) << 3);
            out[(size_t)(b0 + m) * DOUT + nrow] = acc[r] + bo;
        }
    }
}

extern "C" void kernel_launch(void* const* d_in, const int* in_sizes, int n_in,
                              void* d_out, int out_size, void* d_ws, size_t ws_size,
                              hipStream_t stream) {
    (void)in_sizes; (void)n_in; (void)out_size; (void)d_ws; (void)ws_size;
    const float* xs    = (const float*)d_in[0];
    const float* W_ih  = (const float*)d_in[1];
    const float* b_ih  = (const float*)d_in[2];
    const float* W_hh  = (const float*)d_in[3];
    const float* b_hh  = (const float*)d_in[4];
    const float* W_out = (const float*)d_in[5];
    const float* b_out = (const float*)d_in[6];
    float* out = (float*)d_out;

    rnn_scan_wmma<<<dim3(B_ / 16), dim3(256), 0, stream>>>(
        xs, W_ih, b_ih, W_hh, b_hh, W_out, b_out, out);
}
